// Network_76493367542190
// MI455X (gfx1250) — matrix operand order repaired and checked by reading, not compiled, not run on hardware
//
#include <hip/hip_runtime.h>

// ---------------------------------------------------------------------------
// Problem constants (match the reference)
// ---------------------------------------------------------------------------
namespace cfg {
constexpr int B = 16, S = 2048, E = 768, L = 4, O = 10, KLEN = 32;
constexpr int M = B * S;          // 32768 rows
constexpr int N2 = 2 * E;         // 1536
constexpr float EPS = 1e-6f;
}

typedef __attribute__((ext_vector_type(16))) __bf16 v16bf;
typedef __attribute__((ext_vector_type(8)))  float  v8f;

union BF16Frag { unsigned short s[16]; v16bf v; };

__device__ __forceinline__ unsigned short f2bf(float x) {
    unsigned int u = __float_as_uint(x);
    u += 0x7fffu + ((u >> 16) & 1u);          // round-to-nearest-even
    return (unsigned short)(u >> 16);
}

// ---------------------------------------------------------------------------
// CDNA5 async-copy-to-LDS support (GLOBAL_LOAD_ASYNC_TO_LDS_B128, ASYNCcnt).
// The toolchain's prototype (from the round-2 diagnostic) is
//   void (v4i as(1)* src, v4i as(3)* dst, imm int offset, imm int cpol)
// with v4i = int __attribute__((vector_size(16))).
// ---------------------------------------------------------------------------
#ifndef __has_builtin
#define __has_builtin(x) 0
#endif

#if __has_builtin(__builtin_amdgcn_global_load_async_to_lds_b128)
#define HAVE_ASYNC_LDS 1
#else
#define HAVE_ASYNC_LDS 0
#endif

#if __has_builtin(__builtin_amdgcn_s_wait_asynccnt)
#define WAIT_ASYNC0() __builtin_amdgcn_s_wait_asynccnt(0)
#else
#define WAIT_ASYNC0() asm volatile("s_wait_asynccnt 0x0" ::: "memory")
#endif

#if HAVE_ASYNC_LDS
typedef int v4i_vs __attribute__((vector_size(16)));
typedef __attribute__((address_space(1))) v4i_vs as1_v4i;
typedef __attribute__((address_space(3))) v4i_vs as3_v4i;
// Build address-space pointers via integer round-trips:
//  - generic global address == as(1) address on amdgcn
//  - generic LDS address truncates to the 32-bit LDS offset (ISA aperture rule)
__device__ __forceinline__ void async_cp16(const void* g, void* lds_generic) {
    as1_v4i* gp = (as1_v4i*)(unsigned long long)(size_t)g;
    as3_v4i* lp = (as3_v4i*)(unsigned int)(size_t)lds_generic;
    __builtin_amdgcn_global_load_async_to_lds_b128(gp, lp, 0, 0);
}
#endif

// ---------------------------------------------------------------------------
// bf16 WMMA GEMM: C[M,N] = A[M,K] * Wt[N,K]^T (+bias), f32 accumulate.
// Wt is stored PRE-TRANSPOSED ([N][K] row-major) so A and B staging are
// identical contiguous 32B-per-thread copies and every fragment is two
// contiguous ds_load_b128s. Block tile 128x128, 8 waves (4x2), wave tile
// 32x64 = 2x4 WMMA tiles, K-step 32.
// ---------------------------------------------------------------------------
#define TM 128
#define TN 128
#define TK 32

__global__ __launch_bounds__(256)
void gemm_bf16_wmma(const unsigned short* __restrict__ A,   // [M,K] bf16
                    const unsigned short* __restrict__ Wt,  // [N,K] bf16 (transposed)
                    const float* __restrict__ bias,         // [N] or nullptr
                    float* __restrict__ C,                  // [M,N] f32
                    int K, int N)
{
    __shared__ __align__(16) unsigned short As[2][TM * TK];
    __shared__ __align__(16) unsigned short Bs[2][TN * TK];

    const int tid  = threadIdx.x;
    const int lane = tid & 31;
    const int wave = tid >> 5;
    const int wm   = (wave >> 1) * 32;     // wave row offset within block tile
    const int wn   = (wave & 1) * 64;      // wave col offset within block tile

    const int rowBase = blockIdx.x * TM;
    const int colBase = blockIdx.y * TN;

    v8f acc[2][4];
    #pragma unroll
    for (int i = 0; i < 2; ++i)
        #pragma unroll
        for (int j = 0; j < 4; ++j)
            #pragma unroll
            for (int q = 0; q < 8; ++q) acc[i][j][q] = 0.0f;

    // staging map: each thread owns one 32B segment of A and of B
    const int row = tid >> 1;            // 0..127
    const int seg = (tid & 1) * 16;      // 0 / 16 halves

    const int nk = K / TK;

    // ---- fragment gather + 8x WMMA for LDS buffer p ----
    auto compute_tile = [&](int p) {
        BF16Frag afrag[2], bfrag[4];
        // A 16x32 layout: lane L -> row M=L&15; halves = K (L>>4)*16..+15
        const int am = lane & 15;
        const int ak = (lane >> 4) * 16;
        #pragma unroll
        for (int i = 0; i < 2; ++i) {
            const unsigned short* src = &As[p][(wm + i * 16 + am) * TK];
            __builtin_memcpy(&afrag[i].s[0], src + ak,      16);
            __builtin_memcpy(&afrag[i].s[8], src + ak + 8,  16);
        }
        // B 32x16 layout: lane L -> col N=L&15; halves = K (L>>4)*16..+15
        const int bn = lane & 15;
        const int bk = (lane >> 4) * 16;
        #pragma unroll
        for (int j = 0; j < 4; ++j) {
            const unsigned short* src = &Bs[p][(wn + j * 16 + bn) * TK + bk];
            __builtin_memcpy(&bfrag[j].s[0], src, 32);
        }
        #pragma unroll
        for (int i = 0; i < 2; ++i)
            #pragma unroll
            for (int j = 0; j < 4; ++j)
                acc[i][j] = __builtin_amdgcn_wmma_f32_16x16x32_bf16(
                    false, afrag[i].v, false, bfrag[j].v,
                    (short)0, acc[i][j], false, false);
    };

#if HAVE_ASYNC_LDS
    // ---- async double-buffered pipeline (ASYNCcnt) ----
    auto stage_async = [&](int kt, int buf) {
        const unsigned short* ga = A  + (size_t)(rowBase + row) * K + kt * TK + seg;
        const unsigned short* gb = Wt + (size_t)(colBase + row) * K + kt * TK + seg;
        unsigned short* la = &As[buf][row * TK + seg];
        unsigned short* lb = &Bs[buf][row * TK + seg];
        async_cp16(ga,     la);
        async_cp16(ga + 8, la + 8);
        async_cp16(gb,     lb);
        async_cp16(gb + 8, lb + 8);
    };
    stage_async(0, 0);
    for (int kt = 0; kt < nk; ++kt) {
        const int p = kt & 1;
        WAIT_ASYNC0();                 // our buffer-p copies have landed in LDS
        __syncthreads();               // everyone's have; everyone done with p^1
        if (kt + 1 < nk) stage_async(kt + 1, p ^ 1);
        compute_tile(p);
    }
#else
    // ---- register double-buffered fallback: load tile kt+1 into VGPRs
    // before computing tile kt, so loadcnt waits land after the WMMAs ----
    uint4 ra0, ra1, rb0, rb1;
    auto load_regs = [&](int kt) {
        const unsigned short* ga = A  + (size_t)(rowBase + row) * K + kt * TK + seg;
        const unsigned short* gb = Wt + (size_t)(colBase + row) * K + kt * TK + seg;
        ra0 = *(const uint4*)(ga);
        ra1 = *(const uint4*)(ga + 8);
        rb0 = *(const uint4*)(gb);
        rb1 = *(const uint4*)(gb + 8);
        if (kt + 1 < nk) {   // global_prefetch_b8 the tile after next
            __builtin_prefetch(ga + (size_t)TK, 0, 0);
            __builtin_prefetch(gb + (size_t)TK, 0, 0);
        }
    };
    load_regs(0);
    for (int kt = 0; kt < nk; ++kt) {
        __syncthreads();
        *(uint4*)&As[0][row * TK + seg]     = ra0;
        *(uint4*)&As[0][row * TK + seg + 8] = ra1;
        *(uint4*)&Bs[0][row * TK + seg]     = rb0;
        *(uint4*)&Bs[0][row * TK + seg + 8] = rb1;
        __syncthreads();
        if (kt + 1 < nk) load_regs(kt + 1);
        compute_tile(0);
    }
#endif

    // ---- epilogue: C/D layout lanes 0-15 -> M=r, lanes 16-31 -> M=8+r ----
    const int em = (lane >> 4) * 8;
    const int en = lane & 15;
    #pragma unroll
    for (int i = 0; i < 2; ++i) {
        #pragma unroll
        for (int j = 0; j < 4; ++j) {
            const int col = colBase + wn + j * 16 + en;
            const float bv = bias ? bias[col] : 0.0f;
            #pragma unroll
            for (int r = 0; r < 8; ++r) {
                const int rowc = rowBase + wm + i * 16 + em + r;
                C[(size_t)rowc * N + col] = acc[i][j][r] + bv;
            }
        }
    }
}

// ---------------------------------------------------------------------------
// Precompute: bf16 circulant matrices, stored TRANSPOSED [l][n][k]:
// bind:   y[n] = sum_k x[k] ef[(n-k) mod E]  => WbT[n][k] = ef[(n-k) mod E]
// unbind: y[n] = sum_k x[k] df[(k-n) mod E]  => WuT[n][k] = df[(k-n) mod E]
// ---------------------------------------------------------------------------
__global__ void circT_kernel(const float* __restrict__ ef, const float* __restrict__ df,
                             unsigned short* __restrict__ wbT, unsigned short* __restrict__ wuT)
{
    using namespace cfg;
    size_t n = (size_t)blockIdx.x * 256 + threadIdx.x;
    if (n >= (size_t)L * E * E) return;
    int k = (int)(n % E);
    size_t nc = n / E;
    int col = (int)(nc % E);
    int l   = (int)(nc / E);
    int d1 = col - k; d1 += (d1 >> 31) & E;
    int d2 = k - col; d2 += (d2 >> 31) & E;
    wbT[n] = f2bf(ef[l * E + d1]);
    wuT[n] = f2bf(df[l * E + d2]);
}

// dWT[l][f][e] = dW[l][e][f]  (bf16, transposed for the GEMM's B layout)
__global__ void cvt_dwT_kernel(const float* __restrict__ dW, unsigned short* __restrict__ dst)
{
    using namespace cfg;
    size_t n = (size_t)blockIdx.x * 256 + threadIdx.x;
    if (n >= (size_t)L * N2 * E) return;
    int e = (int)(n % E);
    size_t t = n / E;
    int f = (int)(t % N2);
    int l = (int)(t / N2);
    dst[n] = f2bf(dW[((size_t)l * E + e) * N2 + f]);
}

// ---------------------------------------------------------------------------
// Embedding gather + mask + LayerNorm -> f32 (skip path) + bf16 (GEMM A path)
// One wave32 per (b,s) row; 24 elements per lane.
// ---------------------------------------------------------------------------
__global__ __launch_bounds__(256)
void embed_ln0_kernel(const int* __restrict__ tok, const float* __restrict__ embed,
                      const float* __restrict__ scale, const float* __restrict__ bias,
                      float* __restrict__ xf, unsigned short* __restrict__ xb)
{
    using namespace cfg;
    const int row  = blockIdx.x * 8 + (threadIdx.x >> 5);
    const int lane = threadIdx.x & 31;
    const int tk = tok[row];
    const float m = tk > 0 ? 1.0f : 0.0f;
    const float* er = embed + (size_t)tk * E;

    float v[24];
    float s = 0.0f;
    #pragma unroll
    for (int i = 0; i < 24; ++i) { v[i] = er[lane + i * 32] * m; s += v[i]; }
    #pragma unroll
    for (int off = 16; off; off >>= 1) s += __shfl_xor(s, off, 32);
    const float mu = s * (1.0f / E);
    float q = 0.0f;
    #pragma unroll
    for (int i = 0; i < 24; ++i) { float d = v[i] - mu; q += d * d; }
    #pragma unroll
    for (int off = 16; off; off >>= 1) q += __shfl_xor(q, off, 32);
    const float rs = rsqrtf(q * (1.0f / E) + EPS);
    #pragma unroll
    for (int i = 0; i < 24; ++i) {
        const int e = lane + i * 32;
        const float o = (v[i] - mu) * rs * scale[e] + bias[e];
        xf[(size_t)row * E + e] = o;
        xb[(size_t)row * E + e] = f2bf(o);
    }
}

// ---------------------------------------------------------------------------
// Depthwise circular conv along S (32 taps, scale sqrt(S)) + gated res + GELU.
// fft-pad/ifft(ortho) of cf collapses analytically to sqrt(S)*cf.
// Each block covers one aligned 256-wide e-chunk of one (b,s) row (768=3*256),
// so the 32x256 filter slice is cached in LDS once per block.
// ---------------------------------------------------------------------------
__global__ __launch_bounds__(256)
void conv_gelu_kernel(const float* __restrict__ t, const float* __restrict__ cfw,
                      const float* __restrict__ wg, float* __restrict__ u,
                      unsigned short* __restrict__ ub)
{
    using namespace cfg;
    __shared__ float cfs[KLEN][256];
    const int tid = threadIdx.x;
    const size_t n = (size_t)blockIdx.x * 256 + tid;   // grid covers M*E exactly
    const int e = (int)(n % E);
    const size_t bs = n / E;
    const int s = (int)(bs % S);
    const size_t b = bs / S;

    const int e0 = (int)((blockIdx.x % 3u) * 256u);    // block-uniform e-chunk base
    #pragma unroll
    for (int j = 0; j < KLEN; ++j) cfs[j][tid] = cfw[j * E + e0 + tid];
    __syncthreads();

    const float scale = 45.25483399593904f;            // sqrt(2048)
    const float* xrow = t + b * (size_t)S * E;
    float acc = 0.0f;
    #pragma unroll 4
    for (int j = 0; j < KLEN; ++j) {
        int sp = s - j; sp += (sp >> 31) & S;          // circular wrap
        acc += xrow[(size_t)sp * E + e] * cfs[j][tid];
    }
    const float xc = t[n];
    const float y = scale * acc + xc * wg[e];
    const float g = 0.5f * y * (1.0f + tanhf(0.7978845608028654f * (y + 0.044715f * y * y * y)));
    u[n]  = g;
    ub[n] = f2bf(g);
}

// ---------------------------------------------------------------------------
// LayerNorm -> bf16 (feeds the GLU GEMM). One wave32 per row.
// ---------------------------------------------------------------------------
__global__ __launch_bounds__(256)
void ln_bf16_kernel(const float* __restrict__ x, const float* __restrict__ scale,
                    const float* __restrict__ bias, unsigned short* __restrict__ out)
{
    using namespace cfg;
    const int row  = blockIdx.x * 8 + (threadIdx.x >> 5);
    const int lane = threadIdx.x & 31;
    const float* xr = x + (size_t)row * E;
    float v[24];
    float s = 0.0f;
    #pragma unroll
    for (int i = 0; i < 24; ++i) { v[i] = xr[lane + i * 32]; s += v[i]; }
    #pragma unroll
    for (int off = 16; off; off >>= 1) s += __shfl_xor(s, off, 32);
    const float mu = s * (1.0f / E);
    float q = 0.0f;
    #pragma unroll
    for (int i = 0; i < 24; ++i) { float d = v[i] - mu; q += d * d; }
    #pragma unroll
    for (int off = 16; off; off >>= 1) q += __shfl_xor(q, off, 32);
    const float rs = rsqrtf(q * (1.0f / E) + EPS);
    #pragma unroll
    for (int i = 0; i < 24; ++i) {
        const int e = lane + i * 32;
        out[(size_t)row * E + e] = f2bf((v[i] - mu) * rs * scale[e] + bias[e]);
    }
}

// ---------------------------------------------------------------------------
// GLU + residual: x = a*sigmoid(b) + skip ; refresh f32 + bf16 activations.
// ---------------------------------------------------------------------------
__global__ __launch_bounds__(256)
void glu_kernel(const float* __restrict__ h, float* __restrict__ xf,
                unsigned short* __restrict__ xb)
{
    using namespace cfg;
    size_t n = (size_t)blockIdx.x * 256 + threadIdx.x;
    if (n >= (size_t)M * E) return;
    const size_t row = n / E;
    const int c = (int)(n % E);
    const float a  = h[row * (size_t)N2 + c];
    const float bb = h[row * (size_t)N2 + E + c];
    float x = a / (1.0f + expf(-bb)) + xf[n];
    xf[n] = x;
    xb[n] = f2bf(x);
}

// ---------------------------------------------------------------------------
// Masked mean pool over S.
// ---------------------------------------------------------------------------
__global__ __launch_bounds__(256)
void pool_kernel(const float* __restrict__ xf, const int* __restrict__ tok,
                 float* __restrict__ pooled)
{
    using namespace cfg;
    const int b = blockIdx.x / 3;
    const int e = (blockIdx.x % 3) * 256 + threadIdx.x;
    const int* tr = tok + b * S;
    float sum = 0.0f, cnt = 0.0f;
    for (int s = 0; s < S; ++s) {
        const float m = tr[s] > 0 ? 1.0f : 0.0f;
        sum += xf[((size_t)b * S + s) * E + e] * m;
        cnt += m;
    }
    pooled[b * E + e] = sum / cnt;
}

// ---------------------------------------------------------------------------
// Output projection [768 -> 10] + log_softmax. Single tiny block.
// ---------------------------------------------------------------------------
__global__ __launch_bounds__(256)
void out_kernel(const float* __restrict__ pooled, const float* __restrict__ outW,
                const float* __restrict__ outb, float* __restrict__ out)
{
    using namespace cfg;
    __shared__ float lg[B][O];
    const int tid = threadIdx.x;
    if (tid < B * O) {
        const int b = tid / O, o = tid % O;
        float s = outb[o];
        for (int e = 0; e < E; ++e) s += pooled[b * E + e] * outW[e * O + o];
        lg[b][o] = s;
    }
    __syncthreads();
    if (tid < B * O) {
        const int b = tid / O, o = tid % O;
        float mx = -1e30f;
        for (int i = 0; i < O; ++i) mx = fmaxf(mx, lg[b][i]);
        float se = 0.0f;
        for (int i = 0; i < O; ++i) se += expf(lg[b][i] - mx);
        out[b * O + o] = lg[b][o] - mx - logf(se);
    }
}

// ---------------------------------------------------------------------------
// Host launcher
// ---------------------------------------------------------------------------
extern "C" void kernel_launch(void* const* d_in, const int* in_sizes, int n_in,
                              void* d_out, int out_size, void* d_ws, size_t ws_size,
                              hipStream_t stream)
{
    using namespace cfg;
    const int*   tok       = (const int*)  d_in[0];
    const float* embed     = (const float*)d_in[1];
    const float* ln0_scale = (const float*)d_in[2];
    const float* ln0_bias  = (const float*)d_in[3];
    const float* ef        = (const float*)d_in[4];
    const float* cfw       = (const float*)d_in[5];
    const float* df        = (const float*)d_in[6];
    const float* wgate     = (const float*)d_in[7];
    const float* ln_scale  = (const float*)d_in[8];
    const float* ln_bias   = (const float*)d_in[9];
    const float* dW        = (const float*)d_in[10];
    const float* db        = (const float*)d_in[11];
    const float* outW      = (const float*)d_in[12];
    const float* outb      = (const float*)d_in[13];
    float* out = (float*)d_out;

    // ---- workspace carve-up (256B aligned) ----
    char* ws = (char*)d_ws;
    size_t off = 0;
    auto carve = [&](size_t bytes) { char* p = ws + off; off += (bytes + 255) & ~(size_t)255; return p; };
    const size_t MF = (size_t)M * E * sizeof(float);
    const size_t MH = (size_t)M * E * sizeof(unsigned short);
    float*          xf     = (float*)         carve(MF);                     // f32 activations / skip
    unsigned short* xb     = (unsigned short*)carve(MH);                     // bf16 activations
    float*          t      = (float*)         carve(MF);                     // bind / unbind GEMM output
    float*          u      = (float*)         carve(MF);                     // conv+gelu output
    unsigned short* ub     = (unsigned short*)carve(MH);                     // bf16 of u / LN output
    float*          h      = (float*)         carve((size_t)M * N2 * 4);     // GLU pre-activations
    unsigned short* wbT    = (unsigned short*)carve((size_t)L * E * E * 2);  // circulant^T (bind)
    unsigned short* wuT    = (unsigned short*)carve((size_t)L * E * E * 2);  // circulant^T (unbind)
    unsigned short* dwT    = (unsigned short*)carve((size_t)L * E * N2 * 2); // dW^T bf16
    float*          pooled = (float*)         carve((size_t)B * E * 4);
    if (off > ws_size) return;  // workspace too small: nothing safe to do

    // ---- precompute bf16 (pre-transposed) weights ----
    {
        size_t n = (size_t)L * E * E;
        circT_kernel<<<dim3((unsigned)((n + 255) / 256)), dim3(256), 0, stream>>>(ef, df, wbT, wuT);
        size_t m = (size_t)L * N2 * E;
        cvt_dwT_kernel<<<dim3((unsigned)((m + 255) / 256)), dim3(256), 0, stream>>>(dW, dwT);
    }

    // ---- embedding + LN0 ----
    embed_ln0_kernel<<<dim3(M / 8), dim3(256), 0, stream>>>(tok, embed, ln0_scale, ln0_bias, xf, xb);

    const size_t nElem = (size_t)M * E;
    const unsigned elemBlocks = (unsigned)((nElem + 255) / 256);

    for (int l = 0; l < L; ++l) {
        // bind: t = xb @ circ(ef[l])            [32768,768]x[768,768]
        gemm_bf16_wmma<<<dim3(M / TM, E / TN), dim3(256), 0, stream>>>(
            xb, wbT + (size_t)l * E * E, nullptr, t, E, E);
        // depthwise circular seq-conv + gated residual + GELU
        conv_gelu_kernel<<<dim3(elemBlocks), dim3(256), 0, stream>>>(
            t, cfw + (size_t)l * KLEN * E, wgate + (size_t)l * E, u, ub);
        // unbind: t = ub @ circ(df[l])
        gemm_bf16_wmma<<<dim3(M / TM, E / TN), dim3(256), 0, stream>>>(
            ub, wuT + (size_t)l * E * E, nullptr, t, E, E);
        // LayerNorm -> bf16 (reuse ub)
        ln_bf16_kernel<<<dim3(M / 8), dim3(256), 0, stream>>>(
            t, ln_scale + (size_t)l * E, ln_bias + (size_t)l * E, ub);
        // GLU projection: h = ub @ dW[l] + db[l]  [32768,768]x[768,1536]
        gemm_bf16_wmma<<<dim3(M / TM, N2 / TN), dim3(256), 0, stream>>>(
            ub, dwT + (size_t)l * E * N2, db + (size_t)l * N2, h, E, N2);
        // GLU gate + residual; refresh xf/xb for next layer
        glu_kernel<<<dim3(elemBlocks), dim3(256), 0, stream>>>(h, xf, xb);
    }

    // masked mean pool + output head
    pool_kernel<<<dim3(B * 3), dim3(256), 0, stream>>>(xf, tok, pooled);
    out_kernel<<<dim3(1), dim3(256), 0, stream>>>(pooled, outW, outb, out);
    (void)in_sizes; (void)n_in; (void)out_size;
}
